// BERTBlock_13958643712031
// MI455X (gfx1250) — matrix operand order repaired and checked by reading, not compiled, not run on hardware
//
#include <hip/hip_runtime.h>
#include <math.h>

#define BB  8
#define SS  1024
#define EE  1024
#define HHD 16
#define DHD 64
#define HIDD 4096
#define BSROWS (BB * SS)   // 8192

typedef __attribute__((ext_vector_type(16))) __bf16 v16bf;
typedef __attribute__((ext_vector_type(8)))  __bf16 v8bf;
typedef __attribute__((ext_vector_type(8)))  float  v8f;

union BF16x16 { v16bf v; v8bf h[2]; };

// ---------------------------------------------------------------------------
// Tensor Data Mover support (gfx1250): one TDM descriptor moves a whole padded
// 2D tile global->LDS, tracked by TENSORcnt. Guarded so compilation never
// breaks if the builtin is absent; arity differs between clang-22 (5 args,
// ROCm 7.2) and clang-23+ (6 args, therock headers).
// ---------------------------------------------------------------------------
#if defined(__has_builtin)
#if __has_builtin(__builtin_amdgcn_tensor_load_to_lds) && \
    __has_builtin(__builtin_amdgcn_s_wait_tensorcnt)
#define CDNA5_HAS_TDM 1
#endif
#endif

#ifdef CDNA5_HAS_TDM
typedef __attribute__((ext_vector_type(4))) unsigned int u32x4;
typedef __attribute__((ext_vector_type(4))) int          i32x4;
typedef __attribute__((ext_vector_type(8))) int          i32x8;

// 2D tile load: tile_rows rows of tile_k bf16 elements, source row stride
// row_stride elements, destination LDS gets pad_amount dwords of padding every
// pad_interval dwords (codes per D# group1: interval 2^(c+1) dw, amount c+1 dw)
__device__ __forceinline__ void tdm_load_tile_2d(unsigned lds_byte_addr,
                                                 const void* gsrc,
                                                 unsigned tile_k,
                                                 unsigned tile_rows,
                                                 unsigned row_stride,
                                                 unsigned pad_interval_code,
                                                 unsigned pad_amount_code) {
  unsigned long long ga = (unsigned long long)(size_t)gsrc;
  u32x4 g0;
  g0.x = 1u;                                       // count = 1 valid descriptor
  g0.y = lds_byte_addr;                            // LDS dest (bytes)
  g0.z = (unsigned)(ga & 0xFFFFFFFFu);             // global_addr[31:0]
  g0.w = (unsigned)((ga >> 32) & 0x01FFFFFFu)      // global_addr[56:32]
         | (2u << 30);                             // type = 2 ("image")
  const unsigned td0 = 0x7FFFFFFFu;                // huge tensor dims: no OOB
  const unsigned td1 = 0x7FFFFFFFu;
  i32x8 g1;
  g1[0] = (int)((1u << 16)                          // data_size = 2 bytes
                | (1u << 20)                        // pad_enable
                | (pad_interval_code << 22)
                | (pad_amount_code << 25));
  g1[1] = (int)((td0 & 0xFFFFu) << 16);             // tensor_dim0[15:0]
  g1[2] = (int)(((td0 >> 16) & 0xFFFFu) | ((td1 & 0xFFFFu) << 16));
  g1[3] = (int)(((td1 >> 16) & 0xFFFFu) | ((tile_k & 0xFFFFu) << 16)); // tile_dim0
  g1[4] = (int)(tile_rows & 0xFFFFu);               // tile_dim1 (tile_dim2 = 0)
  g1[5] = (int)row_stride;                          // tensor_dim0_stride[31:0]
  g1[6] = 0;
  g1[7] = 0;
  i32x4 z4 = {0, 0, 0, 0};
#if defined(__clang_major__) && (__clang_major__ >= 23)
  i32x8 z8 = {0, 0, 0, 0, 0, 0, 0, 0};
  __builtin_amdgcn_tensor_load_to_lds(g0, g1, z4, z4, z8, 0);
#else
  __builtin_amdgcn_tensor_load_to_lds(g0, g1, z4, z4, 0);
#endif
}

__device__ __forceinline__ unsigned lds_addr_of(const void* p) {
  return (unsigned)(unsigned long long)
      (__attribute__((address_space(3))) const char*)p;
}
#endif  // CDNA5_HAS_TDM

__device__ __forceinline__ unsigned short f32_to_bf16(float f) {
  union { float f; unsigned u; } v; v.f = f;
  unsigned r = v.u + 0x7FFFu + ((v.u >> 16) & 1u);   // round-to-nearest-even
  return (unsigned short)(r >> 16);
}

__device__ __forceinline__ v8f wmma_bf16(v16bf a, v16bf b, v8f c) {
  // D = A(16x32 bf16) x B(32x16 bf16) + C(16x16 f32)
  return __builtin_amdgcn_wmma_f32_16x16x32_bf16(false, a, false, b, (short)0, c,
                                                 false, false);
}

// A fragment (16x32 bf16, MxK): lane holds row = m0+(lane&15),
// K = kb + (lane>>4)*8 .. +7 and +16..+23  (ISA 7.12.2 layout)
__device__ __forceinline__ v16bf frag_a_lds(const unsigned short* lds, int stride,
                                            int m0, int kb, int lane) {
  int row = m0 + (lane & 15);
  int k0  = kb + ((lane >> 4) << 3);
  BF16x16 f;
  f.h[0] = *(const v8bf*)(lds + row * stride + k0);
  f.h[1] = *(const v8bf*)(lds + row * stride + k0 + 16);
  return f.v;
}

// B fragment (32x16 bf16, KxN) from a [N][K] row-major tile:
// lane holds col = n0+(lane&15), K = kb + (lane>>4)*8 .. +7 and +16..+23
__device__ __forceinline__ v16bf frag_b_lds(const unsigned short* lds, int stride,
                                            int n0, int kb, int lane) {
  int col = n0 + (lane & 15);
  int k0  = kb + ((lane >> 4) << 3);
  BF16x16 f;
  f.h[0] = *(const v8bf*)(lds + col * stride + k0);
  f.h[1] = *(const v8bf*)(lds + col * stride + k0 + 16);
  return f.v;
}

// ---------------------------------------------------------------------------
// fp32 -> bf16 conversion
// ---------------------------------------------------------------------------
__global__ __launch_bounds__(256) void cvt_f32_bf16_kernel(
    const float* __restrict__ src, unsigned short* __restrict__ dst, int n) {
  int i = blockIdx.x * 256 + threadIdx.x;
  if (i < n) dst[i] = f32_to_bf16(src[i]);
}

// gate[b*S+s] = mask[b, S-1, s]
__global__ __launch_bounds__(256) void gate_kernel(
    const float* __restrict__ mask, float* __restrict__ gate) {
  int i = blockIdx.x * 256 + threadIdx.x;   // [0, B*S)
  int b = i >> 10, s = i & 1023;
  gate[i] = mask[((size_t)b * SS + (SS - 1)) * SS + s];
}

// ---------------------------------------------------------------------------
// Generic tiled WMMA GEMM: C[M,N] = epi( A[M,K](bf16) * W[N,K](bf16)^T )
// Block tile 128x128, K-step 32, 8 waves (2 in M x 4 in N), wave tile 64x32.
// A/B tiles staged via TDM (tensor_load_to_lds) with hardware padding to a
// 40-half LDS row stride; fallback path uses per-lane b128 loads + prefetch.
// EPI: 0 = store bf16 (QKV)   1 = +resid_f32, store f32 (mh)
//      2 = +bias, GELU, store bf16 (ffn1)
//      3 = (+bias)*gate[row] + resid_f32, store f32 (ffn2)
// ---------------------------------------------------------------------------
template <int EPI>
__global__ __launch_bounds__(256) void gemm_bf16(
    const unsigned short* __restrict__ A, int lda,
    const unsigned short* __restrict__ W, int ldw,
    const float* __restrict__ bias,
    const float* __restrict__ resid,
    const float* __restrict__ gate,
    float* __restrict__ outf,
    unsigned short* __restrict__ outb,
    int M, int N, int K) {
  __shared__ unsigned short ldsA[128 * 40];   // stride 40 halves = 80B (16B mult)
  __shared__ unsigned short ldsB[128 * 40];

  const int tid  = threadIdx.x;
  const int lane = tid & 31;
  const int wave = tid >> 5;
  const int wm   = (wave >> 2) * 64;    // 0 / 64
  const int wn   = (wave & 3) * 32;     // 0,32,64,96
  const int bm   = blockIdx.y * 128;
  const int bn   = blockIdx.x * 128;

  const v8f z = {0.f, 0.f, 0.f, 0.f, 0.f, 0.f, 0.f, 0.f};
  v8f acc[4][2];
#pragma unroll
  for (int i = 0; i < 4; ++i)
#pragma unroll
    for (int j = 0; j < 2; ++j) acc[i][j] = z;

  for (int k0 = 0; k0 < K; k0 += 32) {
    __syncthreads();
#ifdef CDNA5_HAS_TDM
    if (wave == 0) {
      // 128 rows x 32 halves each; pad 4 dwords every 16 dwords -> stride 40
      tdm_load_tile_2d(lds_addr_of(ldsA), A + (size_t)bm * lda + k0,
                       32u, 128u, (unsigned)lda, 3u, 3u);
      tdm_load_tile_2d(lds_addr_of(ldsB), W + (size_t)bn * ldw + k0,
                       32u, 128u, (unsigned)ldw, 3u, 3u);
      __builtin_amdgcn_s_wait_tensorcnt(0);
    }
#else
    // manual staging: 512 chunks of 8 halves each, 2 per thread
#pragma unroll
    for (int i = 0; i < 2; ++i) {
      int c   = tid + i * 256;
      int row = c >> 2;
      int kc  = (c & 3) << 3;
      const unsigned short* ga = A + (size_t)(bm + row) * lda + k0 + kc;
      const unsigned short* gb = W + (size_t)(bn + row) * ldw + k0 + kc;
      *(v8bf*)(ldsA + row * 40 + kc) = *(const v8bf*)ga;
      *(v8bf*)(ldsB + row * 40 + kc) = *(const v8bf*)gb;
      if (k0 + 32 < K) {                      // warm next K tile
        __builtin_prefetch(ga + 32, 0, 3);
        __builtin_prefetch(gb + 32, 0, 3);
      }
    }
#endif
    __syncthreads();

    v16bf af[4], bfv[2];
#pragma unroll
    for (int mi = 0; mi < 4; ++mi) af[mi]  = frag_a_lds(ldsA, 40, wm + mi * 16, 0, lane);
#pragma unroll
    for (int ni = 0; ni < 2; ++ni) bfv[ni] = frag_b_lds(ldsB, 40, wn + ni * 16, 0, lane);
#pragma unroll
    for (int mi = 0; mi < 4; ++mi)
#pragma unroll
      for (int ni = 0; ni < 2; ++ni)
        acc[mi][ni] = wmma_bf16(af[mi], bfv[ni], acc[mi][ni]);
  }

  // epilogue (C layout: VGPR r -> row r (lanes 0-15) / r+8 (lanes 16-31))
  const int hl = lane >> 4, cn = lane & 15;
#pragma unroll
  for (int mi = 0; mi < 4; ++mi)
#pragma unroll
    for (int ni = 0; ni < 2; ++ni)
#pragma unroll
      for (int r = 0; r < 8; ++r) {
        int row    = bm + wm + mi * 16 + r + hl * 8;
        int col    = bn + wn + ni * 16 + cn;
        size_t idx = (size_t)row * N + col;
        float c    = acc[mi][ni][r];
        if (EPI == 0) {
          outb[idx] = f32_to_bf16(c);
        } else if (EPI == 1) {
          outf[idx] = c + resid[idx];
        } else if (EPI == 2) {
          float x = c + bias[col];
          float g = 0.5f * x * (1.0f + erff(x * 0.70710678118f));   // exact GELU
          outb[idx] = f32_to_bf16(g);
        } else {
          outf[idx] = (c + bias[col]) * gate[row] + resid[idx];
        }
      }
}

// ---------------------------------------------------------------------------
// Flash attention with the reference's softmax(s*mask)*mask renormalization.
// Block = (qt, head, batch): 128 query rows, 8 waves x 16 rows each.
// Per key tile (128): S = Q K^T (WMMA), online softmax in registers,
// P staged to LDS (bf16), O += P V (WMMA, accumulate via C operand).
// K tile staged via TDM (padded to 72-half stride); V staged transposed
// manually (TDM cannot transpose). qkv layout: [B*S][3E] bf16.
// ---------------------------------------------------------------------------
__global__ __launch_bounds__(256) void attn_flash(
    const unsigned short* __restrict__ qkv,
    const float* __restrict__ mask,
    unsigned short* __restrict__ hs) {
  __shared__ unsigned short ldsK[128 * 72];       // K tile  [t][d], stride 72
  __shared__ unsigned short ldsV[64 * 136];       // V^T tile [d][t], stride 136
  __shared__ unsigned short ldsP[8 * 16 * 136];   // per-wave P strips [16][128]

  const int b = blockIdx.z, hh = blockIdx.y, qt = blockIdx.x;
  const int lane = threadIdx.x & 31, wave = threadIdx.x >> 5;
  const int hl = lane >> 4, cn = lane & 15;
  const int q0 = qt * 128 + wave * 16;

  // Q fragments straight from global (K-contiguous): 2 k-steps over DH=64
  v16bf qf[2];
  {
    int row = q0 + cn;
    const unsigned short* qrow = qkv + (size_t)(b * SS + row) * (3 * EE) + hh * DHD;
#pragma unroll
    for (int kf = 0; kf < 2; ++kf) {
      int kb = kf * 32 + (hl << 3);
      BF16x16 f;
      f.h[0] = *(const v8bf*)(qrow + kb);
      f.h[1] = *(const v8bf*)(qrow + kb + 16);
      qf[kf] = f.v;
    }
  }

  const v8f z = {0.f, 0.f, 0.f, 0.f, 0.f, 0.f, 0.f, 0.f};
  v8f o[4];
#pragma unroll
  for (int i = 0; i < 4; ++i) o[i] = z;
  float m_r[8], l_r[8];
#pragma unroll
  for (int r = 0; r < 8; ++r) { m_r[r] = -1e30f; l_r[r] = 0.f; }
  unsigned short* pbase = ldsP + wave * 16 * 136;
  const float scale = 0.125f;   // 1/sqrt(64)

  for (int j = 0; j < SS / 128; ++j) {
    __syncthreads();
#ifdef CDNA5_HAS_TDM
    if (wave == 0) {
      // K tile: 128 rows x 64 halves; pad 4 dwords every 32 dwords -> stride 72
      tdm_load_tile_2d(lds_addr_of(ldsK),
                       qkv + (size_t)(b * SS + j * 128) * (3 * EE) + EE + hh * DHD,
                       64u, 128u, 3u * EE, 4u, 3u);
    }
#endif
    // stage transposed V tile: 1024 chunks of 8 halves, 4 per thread
#pragma unroll
    for (int i = 0; i < 4; ++i) {
      int c   = threadIdx.x + i * 256;
      int row = c >> 3;                 // key index within tile
      int kc  = (c & 7) << 3;           // d offset
      const unsigned short* base =
          qkv + (size_t)(b * SS + j * 128 + row) * (3 * EE) + hh * DHD + kc;
#ifndef CDNA5_HAS_TDM
      *(v8bf*)(ldsK + row * 72 + kc) = *(const v8bf*)(base + EE);        // K
#endif
      union { v8bf v; unsigned short e[8]; } vv;
      vv.v = *(const v8bf*)(base + 2 * EE);                               // V
#pragma unroll
      for (int e = 0; e < 8; ++e) ldsV[(kc + e) * 136 + row] = vv.e[e];   // V^T
    }
#ifdef CDNA5_HAS_TDM
    if (wave == 0) __builtin_amdgcn_s_wait_tensorcnt(0);
#endif
    __syncthreads();

    // S strip: 16 rows x 128 keys per wave (8 tiles x 2 k-steps)
    v8f sacc[8];
#pragma unroll
    for (int ni = 0; ni < 8; ++ni) {
      v8f s = z;
      s = wmma_bf16(qf[0], frag_b_lds(ldsK, 72, ni * 16, 0,  lane), s);
      s = wmma_bf16(qf[1], frag_b_lds(ldsK, 72, ni * 16, 32, lane), s);
      sacc[ni] = s;
    }

    // online softmax: s_hat = (s*scale)*mask; rowmax over the 128-wide strip
    float mnew[8];
#pragma unroll
    for (int r = 0; r < 8; ++r) {
      int sq = q0 + r + hl * 8;
      const float* mrow = mask + ((size_t)b * SS + sq) * SS + j * 128 + cn;
      float best = -1e30f;
#pragma unroll
      for (int ni = 0; ni < 8; ++ni) {
        float x = sacc[ni][r] * scale * mrow[ni * 16];
        sacc[ni][r] = x;
        best = fmaxf(best, x);
      }
#pragma unroll
      for (int msk = 1; msk < 16; msk <<= 1)   // reduce within 16-lane half-group
        best = fmaxf(best, __shfl_xor(best, msk, 32));
      mnew[r] = fmaxf(m_r[r], best);
    }
#pragma unroll
    for (int r = 0; r < 8; ++r) {
      float fr = __expf(m_r[r] - mnew[r]);
      l_r[r] *= fr;
      m_r[r] = mnew[r];
#pragma unroll
      for (int t = 0; t < 4; ++t) o[t][r] = o[t][r] * fr;
    }
    // p = exp(s_hat - m) * mask (mask is hot in the 192MB L2, reload is cheap)
#pragma unroll
    for (int r = 0; r < 8; ++r) {
      int sq = q0 + r + hl * 8;
      const float* mrow = mask + ((size_t)b * SS + sq) * SS + j * 128 + cn;
      float sum = 0.f;
#pragma unroll
      for (int ni = 0; ni < 8; ++ni) {
        float p = __expf(sacc[ni][r] - m_r[r]) * mrow[ni * 16];
        sum += p;
        pbase[(r + hl * 8) * 136 + ni * 16 + cn] = f32_to_bf16(p);
      }
#pragma unroll
      for (int msk = 1; msk < 16; msk <<= 1)
        sum += __shfl_xor(sum, msk, 32);
      l_r[r] += sum;
    }
    // O += P V  (P strip is wave-private; per-wave LDS ops are in-order)
#pragma unroll
    for (int ks = 0; ks < 4; ++ks) {
      int kb = ks * 32 + (hl << 3);
      BF16x16 f;
      f.h[0] = *(const v8bf*)(pbase + cn * 136 + kb);
      f.h[1] = *(const v8bf*)(pbase + cn * 136 + kb + 16);
      v16bf pf = f.v;
#pragma unroll
      for (int ni = 0; ni < 4; ++ni) {
        v16bf vf = frag_b_lds(ldsV, 136, ni * 16, ks * 32, lane);
        o[ni] = wmma_bf16(pf, vf, o[ni]);
      }
    }
  }

  // normalize and write head_stack[b,s, h*64+d] (bf16)
#pragma unroll
  for (int ni = 0; ni < 4; ++ni)
#pragma unroll
    for (int r = 0; r < 8; ++r) {
      int sq  = q0 + r + hl * 8;
      float v = o[ni][r] / (l_r[r] + 1e-20f);
      hs[(size_t)(b * SS + sq) * EE + hh * DHD + ni * 16 + cn] = f32_to_bf16(v);
    }
}

// ---------------------------------------------------------------------------
// LayerNorm over E=1024, one block per row, 8 waves
// ---------------------------------------------------------------------------
__global__ __launch_bounds__(256) void layernorm_rows(
    const float* __restrict__ x, const float* __restrict__ g,
    const float* __restrict__ be, float* __restrict__ of,
    unsigned short* __restrict__ ob) {
  const int row = blockIdx.x;
  const float* xr = x + (size_t)row * EE;
  float v[4];
  float s = 0.f;
#pragma unroll
  for (int i = 0; i < 4; ++i) { v[i] = xr[threadIdx.x + i * 256]; s += v[i]; }
  __shared__ float red[8];
#pragma unroll
  for (int m = 1; m < 32; m <<= 1) s += __shfl_xor(s, m, 32);
  if ((threadIdx.x & 31) == 0) red[threadIdx.x >> 5] = s;
  __syncthreads();
  float mu = 0.f;
#pragma unroll
  for (int i = 0; i < 8; ++i) mu += red[i];
  mu *= (1.0f / EE);
  __syncthreads();
  float q = 0.f;
#pragma unroll
  for (int i = 0; i < 4; ++i) { float d = v[i] - mu; q += d * d; }
#pragma unroll
  for (int m = 1; m < 32; m <<= 1) q += __shfl_xor(q, m, 32);
  if ((threadIdx.x & 31) == 0) red[threadIdx.x >> 5] = q;
  __syncthreads();
  float var = 0.f;
#pragma unroll
  for (int i = 0; i < 8; ++i) var += red[i];
  var *= (1.0f / EE);
  float inv = rsqrtf(var + 1e-5f);
#pragma unroll
  for (int i = 0; i < 4; ++i) {
    int c = threadIdx.x + i * 256;
    float y = (v[i] - mu) * inv * g[c] + be[c];
    if (of) of[(size_t)row * EE + c] = y;
    if (ob) ob[(size_t)row * EE + c] = f32_to_bf16(y);
  }
}

// ---------------------------------------------------------------------------
extern "C" void kernel_launch(void* const* d_in, const int* in_sizes, int n_in,
                              void* d_out, int out_size, void* d_ws, size_t ws_size,
                              hipStream_t stream) {
  (void)in_sizes; (void)n_in; (void)out_size; (void)ws_size;
  const float* h     = (const float*)d_in[0];
  const float* mask  = (const float*)d_in[1];
  const float* wq    = (const float*)d_in[2];
  const float* wk    = (const float*)d_in[3];
  const float* wv    = (const float*)d_in[4];
  const float* w_mh  = (const float*)d_in[5];
  const float* g1    = (const float*)d_in[6];
  const float* beta1 = (const float*)d_in[7];
  const float* w1    = (const float*)d_in[8];
  const float* b1    = (const float*)d_in[9];
  const float* w2    = (const float*)d_in[10];
  const float* b2    = (const float*)d_in[11];
  const float* g2    = (const float*)d_in[12];
  const float* beta2 = (const float*)d_in[13];
  float* out = (float*)d_out;

  // workspace carve-up (256B aligned)
  size_t off = 0;
  char* ws = (char*)d_ws;
  auto alloc = [&](size_t bytes) -> void* {
    void* p = ws + off;
    off += (bytes + 255) & ~(size_t)255;
    return p;
  };
  unsigned short* h_bf    = (unsigned short*)alloc((size_t)BSROWS * EE * 2);
  unsigned short* wqkv_bf = (unsigned short*)alloc((size_t)3 * EE * EE * 2);
  unsigned short* wmh_bf  = (unsigned short*)alloc((size_t)EE * EE * 2);
  unsigned short* w1_bf   = (unsigned short*)alloc((size_t)HIDD * EE * 2);
  unsigned short* w2_bf   = (unsigned short*)alloc((size_t)EE * HIDD * 2);
  size_t qkv_bytes  = (size_t)BSROWS * 3 * EE * 2;
  size_t hs_bytes   = (size_t)BSROWS * EE * 2;
  size_t ffn1_bytes = (size_t)BSROWS * HIDD * 2;
  size_t Rbytes = qkv_bytes + hs_bytes;
  if (ffn1_bytes > Rbytes) Rbytes = ffn1_bytes;
  char* R = (char*)alloc(Rbytes);
  unsigned short* qkv  = (unsigned short*)R;                 // live: steps 3-4
  unsigned short* hsb  = (unsigned short*)(R + qkv_bytes);   // live: steps 4-5
  unsigned short* ffn1 = (unsigned short*)R;                 // live: steps 7-8
  float* h2   = (float*)alloc((size_t)BSROWS * EE * 4);      // also hosts pre-LN2
  float* a_f  = (float*)alloc((size_t)BSROWS * EE * 4);
  unsigned short* a_bf = (unsigned short*)alloc((size_t)BSROWS * EE * 2);
  float* gate = (float*)alloc((size_t)BSROWS * 4);

  const int EW = EE * EE;   // 1M elems per head-stacked weight
  // 1) fp32 -> bf16 conversions
  cvt_f32_bf16_kernel<<<(BSROWS * EE + 255) / 256, 256, 0, stream>>>(h, h_bf, BSROWS * EE);
  cvt_f32_bf16_kernel<<<(EW + 255) / 256, 256, 0, stream>>>(wq, wqkv_bf,          EW);
  cvt_f32_bf16_kernel<<<(EW + 255) / 256, 256, 0, stream>>>(wk, wqkv_bf + EW,     EW);
  cvt_f32_bf16_kernel<<<(EW + 255) / 256, 256, 0, stream>>>(wv, wqkv_bf + 2 * EW, EW);
  cvt_f32_bf16_kernel<<<(EW + 255) / 256, 256, 0, stream>>>(w_mh, wmh_bf, EW);
  cvt_f32_bf16_kernel<<<(HIDD * EE + 255) / 256, 256, 0, stream>>>(w1, w1_bf, HIDD * EE);
  cvt_f32_bf16_kernel<<<(EE * HIDD + 255) / 256, 256, 0, stream>>>(w2, w2_bf, EE * HIDD);
  // 2) FFN gate row
  gate_kernel<<<BSROWS / 256, 256, 0, stream>>>(mask, gate);
  // 3) QKV projection: [8192,1024] x [3072,1024]^T -> bf16 [8192,3072]
  gemm_bf16<0><<<dim3(3 * EE / 128, BSROWS / 128), 256, 0, stream>>>(
      h_bf, EE, wqkv_bf, EE, nullptr, nullptr, nullptr, nullptr, qkv,
      BSROWS, 3 * EE, EE);
  // 4) fused attention -> head_stack bf16 [8192,1024]
  attn_flash<<<dim3(SS / 128, HHD, BB), 256, 0, stream>>>(qkv, mask, hsb);
  // 5) mh_layer + residual h -> h2 f32
  gemm_bf16<1><<<dim3(EE / 128, BSROWS / 128), 256, 0, stream>>>(
      hsb, EE, wmh_bf, EE, nullptr, h, nullptr, h2, nullptr,
      BSROWS, EE, EE);
  // 6) attn_norm -> a (f32 + bf16)
  layernorm_rows<<<BSROWS, 256, 0, stream>>>(h2, g1, beta1, a_f, a_bf);
  // 7) FFN1 + bias + GELU -> bf16 [8192,4096]
  gemm_bf16<2><<<dim3(HIDD / 128, BSROWS / 128), 256, 0, stream>>>(
      a_bf, EE, w1_bf, EE, b1, nullptr, nullptr, nullptr, ffn1,
      BSROWS, HIDD, EE);
  // 8) FFN2 + bias, * gate, + residual a -> pre-LN2 f32 (reuse h2)
  gemm_bf16<3><<<dim3(EE / 128, BSROWS / 128), 256, 0, stream>>>(
      ffn1, HIDD, w2_bf, HIDD, b2, a_f, gate, h2, nullptr,
      BSROWS, EE, HIDD);
  // 9) ffn_norm -> output f32
  layernorm_rows<<<BSROWS, 256, 0, stream>>>(h2, g2, beta2, out, nullptr);
}
